// PairformerStack_927712936262
// MI455X (gfx1250) — compile-verified, matrix operand order checked by reading, not run on hardware
//
#include <hip/hip_runtime.h>
#include <math.h>

// ---------------------------------------------------------------------------
// Pairformer stack on gfx1250 (CDNA5). All matmuls via V_WMMA_F32_16X16X32_F16.
// Working set (33.5MB pairwise fp32) is L2-resident (192MB) -> compute bound.
// Strategy: keep every WMMA operand f16 at rest, K-contiguous, so fragments
// are two b128 loads per lane straight from global; no LDS staging in GEMMs.
// ---------------------------------------------------------------------------

#define N_TOK 256
#define DP    128
#define DS    384
#define NN    65536      // N_TOK * N_TOK
#define TRI_H 4
#define TRI_E 32
#define PB_H  16
#define PB_E  64

typedef __attribute__((ext_vector_type(16))) _Float16 v16h;
typedef __attribute__((ext_vector_type(8)))  float    v8f;

union FragH { v16h v; float4 q[2]; };
union H8    { float4 f4; _Float16 h[8]; };

// 16x32 f16 fragment load (A: rows = lane%16; B from [N][K] storage: cols =
// lane%16). Per ISA 7.12.2 lane group 0 holds K {0..7,16..23}, group 1 holds
// K {8..15,24..31} -> two contiguous 16B chunks per lane. 16B alignment kept
// by K being a multiple of 8 everywhere.
__device__ __forceinline__ v16h ld_frag(const _Float16* base, int ld) {
  const int lane = threadIdx.x & 31;
  const _Float16* p = base + (lane & 15) * ld + ((lane >> 4) << 3);
  FragH f;
  f.q[0] = *reinterpret_cast<const float4*>(p);
  f.q[1] = *reinterpret_cast<const float4*>(p + 16);
  return f.v;
}

__device__ __forceinline__ v8f wmma_f16(v16h a, v16h b, v8f c) {
  return __builtin_amdgcn_wmma_f32_16x16x32_f16(false, a, false, b, (short)0, c,
                                                false, false);
}

__device__ __forceinline__ float sigm(float x) { return 1.f / (1.f + __expf(-x)); }

// ---------------------------------------------------------------------------
// Weight transpose+convert: W[K][N] f32 -> Wt[Npad][K] f16 (pad rows zeroed),
// so GEMM B-fragments are direct K-contiguous global loads with no masking.
// ---------------------------------------------------------------------------
__global__ void k_wconv(const float* __restrict__ W, _Float16* __restrict__ Wt,
                        int K, int N, int Npad)
{
  int idx = blockIdx.x * 256 + threadIdx.x;
  if (idx >= Npad * K) return;
  int n = idx / K, k = idx % K;
  Wt[idx] = (_Float16)((n < N) ? W[(long)k * N + n] : 0.f);
}

// ---------------------------------------------------------------------------
// LDS-free GEMM: Y[M,N] = X[M,K]_f16 @ Wt[Npad,K]_f16^T.
// Block = 128 threads (4 waves, 2x2); wave tile 32x32 (4 accumulators).
// Per K-step: 8 global b128 loads + 4 WMMA. M mult of 64, K mult of 32.
// ---------------------------------------------------------------------------
__global__ __launch_bounds__(128) void k_gemm(
    const _Float16* __restrict__ X, const _Float16* __restrict__ Wt,
    float* __restrict__ Yf, _Float16* __restrict__ Yh,
    int N, int K, int ldy)
{
  const int wave = threadIdx.x >> 5;
  const int m0 = blockIdx.y * 64 + (wave >> 1) * 32;
  const int n0 = blockIdx.x * 64 + (wave & 1) * 32;
  const _Float16* Xb = X  + (long)m0 * K;
  const _Float16* Wb = Wt + (long)n0 * K;
  v8f c00 = {}, c01 = {}, c10 = {}, c11 = {};
  for (int k0 = 0; k0 < K; k0 += 32) {
    v16h a0 = ld_frag(Xb + k0, K);
    v16h a1 = ld_frag(Xb + 16 * K + k0, K);
    v16h b0 = ld_frag(Wb + k0, K);
    v16h b1 = ld_frag(Wb + 16 * K + k0, K);
    c00 = wmma_f16(a0, b0, c00);
    c01 = wmma_f16(a0, b1, c01);
    c10 = wmma_f16(a1, b0, c10);
    c11 = wmma_f16(a1, b1, c11);
  }
  const int lane = threadIdx.x & 31;
  const int crow = (lane >> 4) << 3;
  const int ccol = lane & 15;
  auto st = [&](v8f c, int mm, int nn) {
    if (nn + ccol < N) {
#pragma unroll
      for (int i = 0; i < 8; ++i) {
        long off = (long)(mm + crow + i) * ldy + nn + ccol;
        if (Yf) Yf[off] = c[i];
        if (Yh) Yh[off] = (_Float16)c[i];
      }
    }
  };
  st(c00, m0, n0);      st(c01, m0, n0 + 16);
  st(c10, m0 + 16, n0); st(c11, m0 + 16, n0 + 16);
}

// ---------------------------------------------------------------------------
// Row LayerNorm -> f16 output (all LN consumers are WMMA GEMMs).
// One wave per row, wave32 shuffle reduction.
// ---------------------------------------------------------------------------
__global__ void k_ln(const float* __restrict__ X, const float* __restrict__ g,
                     const float* __restrict__ b, _Float16* __restrict__ Y,
                     int rows, int D)
{
  int row = blockIdx.x * (blockDim.x >> 5) + (threadIdx.x >> 5);
  if (row >= rows) return;
  int lane = threadIdx.x & 31;
  const float* x = X + (long)row * D;
  float s = 0.f, s2 = 0.f;
  for (int i = lane; i < D; i += 32) { float v = x[i]; s += v; s2 += v * v; }
#pragma unroll
  for (int o = 16; o; o >>= 1) { s += __shfl_xor(s, o, 32); s2 += __shfl_xor(s2, o, 32); }
  float mean = s / D;
  float var  = s2 / D - mean * mean;
  float inv  = rsqrtf(var + 1e-5f);
  _Float16* y = Y + (long)row * D;
  for (int i = lane; i < D; i += 32)
    y[i] = (_Float16)((x[i] - mean) * inv * g[i] + b[i]);
}

// ---------------------------------------------------------------------------
// Triangle mult: GLU + repack chunk into d-major f16 tiles A[d][r][c] (and B),
// optional (r,c) transpose for the "incoming" einsum orientation.
// ---------------------------------------------------------------------------
__global__ void k_glu_repack(const float* __restrict__ H, _Float16* __restrict__ A,
                             _Float16* __restrict__ B, int r0, int rows, int transpose)
{
  int idx = blockIdx.x * blockDim.x + threadIdx.x;
  if (idx >= rows * DP) return;
  int lr = idx / DP, d = idx % DP;
  int p = r0 + lr;
  int i = p >> 8, j = p & 255;
  const float* h = H + (long)lr * (4 * DP);
  float lv = h[d]      * sigm(h[2 * DP + d]);
  float rv = h[DP + d] * sigm(h[3 * DP + d]);
  long dst = transpose ? ((long)d * NN + (long)j * N_TOK + i)
                       : ((long)d * NN + (long)i * N_TOK + j);
  A[dst] = (_Float16)lv;
  B[dst] = (_Float16)rv;
}

// Out[i][j][d] = sum_k A[d][i][k]*B[d][j][k]  (128 independent 256^3 GEMMs).
// Wave tile 32x32: per K-step 4 b128-frag loads + 4 WMMA, direct from global.
__global__ __launch_bounds__(128) void k_tri_einsum(
    const _Float16* __restrict__ A, const _Float16* __restrict__ B,
    float* __restrict__ O)
{
  const int d  = blockIdx.z;
  const int i0 = blockIdx.x * 32;                           // 0..7 tiles
  const int j0 = (blockIdx.y * 4 + (threadIdx.x >> 5)) * 32; // 0..7 tiles
  const _Float16* Ab = A + (long)d * NN + (long)i0 * N_TOK;
  const _Float16* Bb = B + (long)d * NN + (long)j0 * N_TOK;
  v8f c00 = {}, c01 = {}, c10 = {}, c11 = {};
#pragma unroll
  for (int k0 = 0; k0 < N_TOK; k0 += 32) {
    v16h a0 = ld_frag(Ab + k0, N_TOK);
    v16h a1 = ld_frag(Ab + 16 * N_TOK + k0, N_TOK);
    v16h b0 = ld_frag(Bb + k0, N_TOK);
    v16h b1 = ld_frag(Bb + 16 * N_TOK + k0, N_TOK);
    c00 = wmma_f16(a0, b0, c00);
    c01 = wmma_f16(a0, b1, c01);
    c10 = wmma_f16(a1, b0, c10);
    c11 = wmma_f16(a1, b1, c11);
  }
  const int lane = threadIdx.x & 31;
  const int crow = (lane >> 4) << 3;
  const int ccol = lane & 15;
  auto st = [&](v8f c, int ib, int jb) {
#pragma unroll
    for (int q = 0; q < 8; ++q)
      O[((long)(ib + crow + q) * N_TOK + jb + ccol) * DP + d] = c[q];
  };
  st(c00, i0, j0);      st(c01, i0, j0 + 16);
  st(c10, i0 + 16, j0); st(c11, i0 + 16, j0 + 16);
}

// ---------------------------------------------------------------------------
// Fused attention, templated on head dim E (32 triangle, 64 pair-bias).
// 128 threads (4 waves) x 16 query rows. K/V staged via b128 loads; QK^T and
// PV are WMMA; bias+scale+softmax fp32 in LDS. Strides encode the transpose.
// ---------------------------------------------------------------------------
template <int E>
__global__ __launch_bounds__(128) void k_attn(
    const _Float16* __restrict__ Q, const _Float16* __restrict__ K,
    const _Float16* __restrict__ V, float* __restrict__ O,
    const float* __restrict__ bias,
    long rstride, long tstride, int hstride, long bqs, long bks, float scale)
{
  extern __shared__ char smem[];
  _Float16* Ks = (_Float16*)smem;             // [256][E]
  _Float16* Vt = Ks + 256 * E;                // [E][256]
  _Float16* Ps = Vt + 256 * E;                // [4][16][256]
  float*    Ls = (float*)(Ps + 4 * 16 * 256); // [4][16][256]

  const int wave = threadIdx.x >> 5;
  const int lane = threadIdx.x & 31;
  const int r = blockIdx.y, h = blockIdx.z;
  const long base = (long)r * rstride + (long)h * hstride;

  constexpr int CH = E / 8;   // 8-half chunks per row
  for (int idx = threadIdx.x; idx < 256 * CH; idx += 128) {
    int t = idx / CH, c8 = (idx % CH) * 8;
    H8 kk; kk.f4 = *reinterpret_cast<const float4*>(K + base + (long)t * tstride + c8);
    *reinterpret_cast<float4*>(Ks + t * E + c8) = kk.f4;
    H8 vv; vv.f4 = *reinterpret_cast<const float4*>(V + base + (long)t * tstride + c8);
#pragma unroll
    for (int e = 0; e < 8; ++e) Vt[(c8 + e) * 256 + t] = vv.h[e];
  }
  __syncthreads();

  const int q0 = blockIdx.x * 64 + wave * 16;
  float*    L = Ls + wave * 16 * 256;
  _Float16* P = Ps + wave * 16 * 256;

  v16h aq[E / 32];
#pragma unroll
  for (int ks = 0; ks < E / 32; ++ks)
    aq[ks] = ld_frag(Q + base + (long)q0 * tstride + ks * 32, (int)tstride);

  for (int ct = 0; ct < 16; ++ct) {
    v8f acc = {};
#pragma unroll
    for (int ks = 0; ks < E / 32; ++ks)
      acc = wmma_f16(aq[ks], ld_frag(Ks + ct * 16 * E + ks * 32, E), acc);
    const int n  = ct * 16 + (lane & 15);
    const int mb = (lane >> 4) << 3;
#pragma unroll
    for (int i = 0; i < 8; ++i)
      L[(mb + i) * 256 + n] =
          acc[i] * scale + bias[(long)(q0 + mb + i) * bqs + (long)n * bks + h];
  }

  for (int m = 0; m < 16; ++m) {   // wave-private softmax over 256 keys
    float mx = -1e30f;
    for (int c = lane; c < 256; c += 32) mx = fmaxf(mx, L[m * 256 + c]);
#pragma unroll
    for (int o = 16; o; o >>= 1) mx = fmaxf(mx, __shfl_xor(mx, o, 32));
    float s = 0.f;
    for (int c = lane; c < 256; c += 32) {
      float e = __expf(L[m * 256 + c] - mx);
      L[m * 256 + c] = e; s += e;
    }
#pragma unroll
    for (int o = 16; o; o >>= 1) s += __shfl_xor(s, o, 32);
    float inv = 1.f / s;
    for (int c = lane; c < 256; c += 32)
      P[m * 256 + c] = (_Float16)(L[m * 256 + c] * inv);
  }

#pragma unroll
  for (int et = 0; et < E / 16; ++et) {
    v8f acc = {};
#pragma unroll
    for (int k0 = 0; k0 < 256; k0 += 32)
      acc = wmma_f16(ld_frag(P + k0, 256), ld_frag(Vt + et * 16 * 256 + k0, 256), acc);
    const int e  = et * 16 + (lane & 15);
    const int mb = (lane >> 4) << 3;
#pragma unroll
    for (int i = 0; i < 8; ++i)
      O[base + (long)(q0 + mb + i) * tstride + e] = acc[i];
  }
}

// ---------------------------------------------------------------------------
// Elementwise helpers
// ---------------------------------------------------------------------------
__global__ void k_add(float* __restrict__ A, const float* __restrict__ B, long n) {
  long i = (long)blockIdx.x * blockDim.x + threadIdx.x;
  if (i < n) A[i] += B[i];
}
__global__ void k_gate_add(float* __restrict__ A, const float* __restrict__ T,
                           const float* __restrict__ G, long n) {
  long i = (long)blockIdx.x * blockDim.x + threadIdx.x;
  if (i < n) A[i] += T[i] * sigm(G[i]);
}
__global__ void k_gate_mul16(const float* __restrict__ O, const float* __restrict__ G,
                             _Float16* __restrict__ U, long n) {
  long i = (long)blockIdx.x * blockDim.x + threadIdx.x;
  if (i < n) U[i] = (_Float16)(O[i] * sigm(G[i]));
}
__global__ void k_swiglu(const float* __restrict__ H, _Float16* __restrict__ U,
                         long n, int Dh) {
  long i = (long)blockIdx.x * blockDim.x + threadIdx.x;
  if (i >= n) return;
  long row = i / Dh; int c = (int)(i % Dh);
  float a = H[row * (2L * Dh) + c];
  float g = H[row * (2L * Dh) + Dh + c];
  U[i] = (_Float16)(a * sigm(a) * g);   // silu(a) * g
}

// ---------------------------------------------------------------------------
// Host orchestration
// ---------------------------------------------------------------------------
extern "C" void kernel_launch(void* const* d_in, const int* in_sizes, int n_in,
                              void* d_out, int out_size, void* d_ws, size_t ws_size,
                              hipStream_t stream) {
  (void)in_sizes; (void)n_in; (void)out_size; (void)ws_size;

  char* wp = (char*)d_ws;
  auto alloc = [&](size_t bytes) -> char* {
    char* p = wp; wp += (bytes + 255) & ~(size_t)255; return p;
  };
  float*    PW   = (float*)alloc((size_t)NN * DP * 4);      // pairwise state
  float*    S    = (float*)alloc((size_t)N_TOK * DS * 4);   // single state
  _Float16* SH   = (_Float16*)alloc((size_t)N_TOK * 1024 * 2); // single-side f16
  _Float16* XH   = (_Float16*)alloc((size_t)NN * DP * 2);   // pair-side f16
  float*    G    = (float*)alloc((size_t)NN * DP * 4);      // raw gates
  float*    O    = (float*)alloc((size_t)NN * DP * 4);      // einsum/attn out
  float*    T    = (float*)alloc((size_t)NN * DP * 4);      // GEMM f32 out
  float*    H    = (float*)alloc((size_t)8192 * 1024 * 4);  // chunk hidden
  _Float16* HU   = (_Float16*)alloc((size_t)8192 * 512 * 2);// swiglu f16 out
  _Float16* FA   = (_Float16*)alloc((size_t)NN * DP * 2);   // q / tri A
  _Float16* FB   = (_Float16*)alloc((size_t)NN * DP * 2);   // k / tri B
  _Float16* FC   = (_Float16*)alloc((size_t)NN * DP * 2);   // v
  _Float16* WT   = (_Float16*)alloc((size_t)3072 * 1536 * 2); // transposed weights
  float*    BIAS = (float*)alloc((size_t)NN * PB_H * 4);

  hipMemcpyAsync(S,  d_in[0], (size_t)N_TOK * DS * 4, hipMemcpyDeviceToDevice, stream);
  hipMemcpyAsync(PW, d_in[1], (size_t)NN * DP * 4,    hipMemcpyDeviceToDevice, stream);

  int cur = 2;
  auto nx = [&]() -> const float* { return (const float*)d_in[cur++]; };

  // convert weight then run LDS-free WMMA GEMM
  auto gemm = [&](const _Float16* X, const float* W, float* Yf, _Float16* Yh,
                  int M, int N, int K, int ldy) {
    int Npad = (N + 63) & ~63;
    k_wconv<<<(Npad * K + 255) / 256, 256, 0, stream>>>(W, WT, K, N, Npad);
    k_gemm<<<dim3(Npad / 64, M / 64), 128, 0, stream>>>(X, WT, Yf, Yh, N, K, ldy);
  };
  auto ln = [&](const float* X, const float* g, const float* b, _Float16* Y,
                int rows, int D) {
    k_ln<<<(rows + 7) / 8, 256, 0, stream>>>(X, g, b, Y, rows, D);
  };

  auto tri_mult = [&](bool outgoing) {
    const float *lng = nx(), *lnb = nx(), *wlr = nx(), *wgate = nx();
    const float *log_ = nx(), *lob = nx(), *wout = nx();
    ln(PW, lng, lnb, XH, NN, DP);
    gemm(XH, wgate, G, nullptr, NN, DP, DP, DP);
    for (int ch = 0; ch < 8; ++ch) {                 // chunk 8192 rows
      int r0 = ch * 8192;
      gemm(XH + (long)r0 * DP, wlr, H, nullptr, 8192, 4 * DP, DP, 4 * DP);
      if (outgoing)  // A <- left[i][k], B <- right[j][k]
        k_glu_repack<<<(8192 * DP) / 256, 256, 0, stream>>>(H, FA, FB, r0, 8192, 0);
      else           // A <- right[k][i]^T, B <- left[k][j]^T (swap + transpose)
        k_glu_repack<<<(8192 * DP) / 256, 256, 0, stream>>>(H, FB, FA, r0, 8192, 1);
    }
    k_tri_einsum<<<dim3(8, 2, DP), 128, 0, stream>>>(FA, FB, O);
    ln(O, log_, lob, XH, NN, DP);
    gemm(XH, wout, T, nullptr, NN, DP, DP, DP);
    k_gate_add<<<(NN * DP) / 256, 256, 0, stream>>>(PW, T, G, (long)NN * DP);
  };

  auto tri_attn = [&](bool transpose) {
    const float *lng = nx(), *lnb = nx(), *wbias = nx();
    const float *wq = nx(), *wk = nx(), *wv = nx(), *wg = nx(), *wo = nx();
    ln(PW, lng, lnb, XH, NN, DP);
    gemm(XH, wbias, BIAS, nullptr, NN, TRI_H, DP, TRI_H);
    gemm(XH, wq, nullptr, FA, NN, DP, DP, DP);
    gemm(XH, wk, nullptr, FB, NN, DP, DP, DP);
    gemm(XH, wv, nullptr, FC, NN, DP, DP, DP);
    gemm(XH, wg, G, nullptr, NN, DP, DP, DP);
    long rs  = transpose ? DP : (long)N_TOK * DP;
    long ts  = transpose ? (long)N_TOK * DP : DP;
    long bqs = transpose ? TRI_H : (long)N_TOK * TRI_H;
    long bks = transpose ? (long)N_TOK * TRI_H : TRI_H;
    size_t smem = (size_t)256 * TRI_E * 2 * 2 + 4 * 16 * 256 * 2 + 4 * 16 * 256 * 4;
    k_attn<TRI_E><<<dim3(4, N_TOK, TRI_H), 128, smem, stream>>>(
        FA, FB, FC, O, BIAS, rs, ts, TRI_E, bqs, bks, 0.17677669529f);
    k_gate_mul16<<<(NN * DP) / 256, 256, 0, stream>>>(O, G, XH, (long)NN * DP);
    gemm(XH, wo, T, nullptr, NN, DP, DP, DP);
    k_add<<<(NN * DP) / 256, 256, 0, stream>>>(PW, T, (long)NN * DP);
  };

  auto pair_transition = [&]() {
    const float *lng = nx(), *lnb = nx(), *w1 = nx(), *w2 = nx();
    ln(PW, lng, lnb, XH, NN, DP);
    for (int ch = 0; ch < 8; ++ch) {
      long r0 = (long)ch * 8192;
      gemm(XH + r0 * DP, w1, H, nullptr, 8192, 8 * DP, DP, 8 * DP);
      k_swiglu<<<(8192 * 4 * DP) / 256, 256, 0, stream>>>(H, HU, (long)8192 * 4 * DP, 4 * DP);
      gemm(HU, w2, T + r0 * DP, nullptr, 8192, DP, 4 * DP, DP);
    }
    k_add<<<(NN * DP) / 256, 256, 0, stream>>>(PW, T, (long)NN * DP);
  };

  auto pair_bias_attn = [&]() {
    const float *lsg = nx(), *lsb = nx(), *lpg = nx(), *lpb = nx(), *wbias = nx();
    const float *wq = nx(), *wk = nx(), *wv = nx(), *wg = nx(), *wo = nx();
    const int inner = PB_H * PB_E;   // 1024
    ln(S, lsg, lsb, SH, N_TOK, DS);
    ln(PW, lpg, lpb, XH, NN, DP);
    gemm(XH, wbias, BIAS, nullptr, NN, PB_H, DP, PB_H);
    gemm(SH, wq, nullptr, FA, N_TOK, inner, DS, inner);
    gemm(SH, wk, nullptr, FB, N_TOK, inner, DS, inner);
    gemm(SH, wv, nullptr, FC, N_TOK, inner, DS, inner);
    gemm(SH, wg, G, nullptr, N_TOK, inner, DS, inner);
    size_t smem = (size_t)256 * PB_E * 2 * 2 + 4 * 16 * 256 * 2 + 4 * 16 * 256 * 4;
    k_attn<PB_E><<<dim3(4, 1, PB_H), 128, smem, stream>>>(
        FA, FB, FC, O, BIAS, 0L, (long)inner, PB_E,
        (long)N_TOK * PB_H, (long)PB_H, 0.125f);
    k_gate_mul16<<<(N_TOK * inner) / 256, 256, 0, stream>>>(O, G, SH, (long)N_TOK * inner);
    gemm(SH, wo, T, nullptr, N_TOK, DS, inner, DS);
    k_add<<<(N_TOK * DS + 255) / 256, 256, 0, stream>>>(S, T, (long)N_TOK * DS);
  };

  auto single_transition = [&]() {
    const float *lng = nx(), *lnb = nx(), *w1 = nx(), *w2 = nx();
    ln(S, lng, lnb, SH, N_TOK, DS);
    gemm(SH, w1, H, nullptr, N_TOK, 8 * DS, DS, 8 * DS);
    k_swiglu<<<(N_TOK * 4 * DS) / 256, 256, 0, stream>>>(H, HU, (long)N_TOK * 4 * DS, 4 * DS);
    gemm(HU, w2, T, nullptr, N_TOK, DS, 4 * DS, DS);
    k_add<<<(N_TOK * DS + 255) / 256, 256, 0, stream>>>(S, T, (long)N_TOK * DS);
  };

  for (int layer = 0; layer < 2; ++layer) {
    tri_mult(true);        // tm_out
    tri_mult(false);       // tm_in
    tri_attn(false);       // ta_start
    tri_attn(true);        // ta_end
    pair_transition();     // pair_tr
    pair_bias_attn();      // pb
    single_transition();   // single_tr
  }

  // outputs: (s, pw) concatenated flat
  hipMemcpyAsync(d_out, S, (size_t)N_TOK * DS * 4, hipMemcpyDeviceToDevice, stream);
  hipMemcpyAsync((float*)d_out + N_TOK * DS, PW, (size_t)NN * DP * 4,
                 hipMemcpyDeviceToDevice, stream);
}
